// LinformerHead_18683107737674
// MI455X (gfx1250) — matrix operand order repaired and checked by reading, not compiled, not run on hardware
//
#include <hip/hip_runtime.h>

// Model dims
#define BB     16
#define SEQ    4096
#define DIM    512
#define NH     8
#define DHD    64
#define KPROJ  256
#define NLAYER 4
#define FFDIM  2048
#define NCLS   10
#define CHROWS 8192   // FFN row chunk

typedef __attribute__((ext_vector_type(16))) __bf16 v16bf;
typedef __attribute__((ext_vector_type(8)))  __bf16 v8bf;
typedef __attribute__((ext_vector_type(8)))  float  v8f;

// ---------------- helpers ----------------

__device__ __forceinline__ __bf16 f2bf(float f) {
    unsigned u = __builtin_bit_cast(unsigned, f);
    u += 0x7FFFu + ((u >> 16) & 1u);            // round-to-nearest-even
    unsigned short s = (unsigned short)(u >> 16);
    return __builtin_bit_cast(__bf16, s);
}

__device__ __forceinline__ v16bf cat8(v8bf lo, v8bf hi) {
    return __builtin_shufflevector(lo, hi, 0, 1, 2, 3, 4, 5, 6, 7,
                                           8, 9, 10, 11, 12, 13, 14, 15);
}

// A fragment, 16x32 bf16, A row-major (k contiguous), element (m,k)=A[m*rs+k].
// lane<16: m=lane, k in {0..7,16..23}; lane>=16: same m, k+8.
// Two 16-byte contiguous loads per lane.
__device__ __forceinline__ v16bf load_a_rm(const __bf16* __restrict__ A,
                                           long rs, int lane) {
    const int g = lane >> 4, m = lane & 15;
    const __bf16* base = A + (long)m * rs + g * 8;
    return cat8(*(const v8bf*)base, *(const v8bf*)(base + 16));
}

// B fragment, 32x16 bf16, B k-contiguous per column: element (k,n)=B[n*cs+k].
// lane<16: n=lane, k in {0..7,16..23}; lane>=16: same n, k+8.
// Two 16-byte contiguous loads per lane.
__device__ __forceinline__ v16bf load_b_kc(const __bf16* __restrict__ B,
                                           long cs, int lane) {
    const int g = lane >> 4, n = lane & 15;
    const __bf16* base = B + (long)n * cs + g * 8;
    return cat8(*(const v8bf*)base, *(const v8bf*)(base + 16));
}

#define WMMA_BF16(a, b, c) \
    __builtin_amdgcn_wmma_f32_16x16x32_bf16(false, (a), false, (b), (short)0, (c), false, false)

// ---------------- kernels ----------------

// fused fp32 -> bf16 convert + transpose: src (L,R,C) -> dst (L,C,R)
__global__ void k_cvt_t(const float* __restrict__ src, __bf16* __restrict__ dst,
                        int R, int C, long total) {
    long i = (long)blockIdx.x * blockDim.x + threadIdx.x;
    if (i >= total) return;
    long rc  = (long)R * C;
    long l   = i / rc, rem = i - l * rc;
    int  r   = (int)(rem / C), c = (int)(rem - (long)r * C);
    dst[l * rc + (long)c * R + r] = f2bf(src[i]);
}

__global__ void k_embed(const int* __restrict__ x, const float* __restrict__ emb,
                        const float* __restrict__ pos, float* __restrict__ h) {
    long i = (long)blockIdx.x * blockDim.x + threadIdx.x;
    const long total = (long)BB * SEQ * DIM;
    if (i >= total) return;
    int  d  = (int)(i % DIM);
    long bn = i / DIM;
    int  n  = (int)(bn % SEQ);
    int  tok = x[bn];
    h[i] = emb[(long)tok * DIM + d] + pos[(long)n * DIM + d];
}

// LayerNorm over D=512, one 256-thread block per row, bf16 output
__global__ void k_ln(const float* __restrict__ h, const float* __restrict__ gmm,
                     const float* __restrict__ bta, __bf16* __restrict__ y) {
    __shared__ float s[256];
    const int  tid = threadIdx.x;
    const long row = blockIdx.x;
    const float* x = h + row * DIM;
    float v0 = x[tid], v1 = x[tid + 256];
    s[tid] = v0 + v1;
    __syncthreads();
    for (int off = 128; off; off >>= 1) { if (tid < off) s[tid] += s[tid + off]; __syncthreads(); }
    float mean = s[0] * (1.0f / DIM);
    __syncthreads();
    float d0 = v0 - mean, d1 = v1 - mean;
    s[tid] = d0 * d0 + d1 * d1;
    __syncthreads();
    for (int off = 128; off; off >>= 1) { if (tid < off) s[tid] += s[tid + off]; __syncthreads(); }
    float r = rsqrtf(s[0] * (1.0f / DIM) + 1e-5f);
    y[row * DIM + tid]       = f2bf(d0 * r * gmm[tid]       + bta[tid]);
    y[row * DIM + tid + 256] = f2bf(d1 * r * gmm[tid + 256] + bta[tid + 256]);
}

// kv (B,K,DHD) -> kvT (B,DHD,K)
__global__ void k_kvT(const __bf16* __restrict__ kv, __bf16* __restrict__ kvT) {
    long i = (long)blockIdx.x * blockDim.x + threadIdx.x;
    const long total = (long)BB * KPROJ * DHD;
    if (i >= total) return;
    long b  = i / (KPROJ * DHD), rem = i - b * (KPROJ * DHD);
    int  k  = (int)(rem / DHD), d = (int)(rem - (long)k * DHD);
    kvT[b * (KPROJ * DHD) + (long)d * KPROJ + k] = kv[i];
}

// Generic bf16 WMMA GEMM: C(m,nc) = sum_k A(m,k)*B(k,nc) [+bias] [gelu] [+resid]
//   A row-major (k contiguous): A[z*sa_b + m*sa_row + k]
//   B k-contiguous per column:  B[z*sb_b + nc*sb_col + k]
//   f32 store/resid row-major ldc at offset z*sc_b; bf16 store strides (scb_m, scb_n)
// One wave computes a 16x64 strip; block = (32,4) -> 64 rows.
__global__ void k_gemm(const __bf16* __restrict__ A, const __bf16* __restrict__ Bm,
                       const float* __restrict__ bias, const float* __restrict__ resid,
                       float* __restrict__ Cf, __bf16* __restrict__ Cb,
                       int M, int Kk,
                       long sa_row, long sa_b, long sb_col, long sb_b,
                       int ldc, long scb_m, long scb_n, long sc_b, int gelu_flag) {
    const int lane = threadIdx.x;
    const int m0 = (blockIdx.y * 4 + threadIdx.y) * 16;
    const int n0 = blockIdx.x * 64;
    const int z  = blockIdx.z;
    if (m0 >= M) return;

    const __bf16* Ab = A + (long)z * sa_b + (long)m0 * sa_row;
    const __bf16* Bb = Bm + (long)z * sb_b + (long)n0 * sb_col;
    const long coff = (long)z * sc_b;

    v8f acc[4] = {};
    for (int k0 = 0; k0 < Kk; k0 += 32) {
        v16bf a = load_a_rm(Ab + k0, sa_row, lane);
#pragma unroll
        for (int t = 0; t < 4; ++t) {
            v16bf b = load_b_kc(Bb + (long)(t * 16) * sb_col + k0, sb_col, lane);
            acc[t] = WMMA_BF16(a, b, acc[t]);
        }
    }

    const int nlane = lane & 15, mh = (lane >> 4) * 8;
#pragma unroll
    for (int t = 0; t < 4; ++t) {
#pragma unroll
        for (int r = 0; r < 8; ++r) {
            int m  = m0 + mh + r;
            int nc = n0 + t * 16 + nlane;
            float v = acc[t][r];
            if (bias)      v += bias[nc];
            if (gelu_flag) v = 0.5f * v * (1.0f + erff(v * 0.70710678118654752f));
            if (resid) v += resid[coff + (long)m * ldc + nc];
            if (Cf) Cf[coff + (long)m * ldc + nc] = v;
            if (Cb) Cb[coff + (long)m * scb_m + (long)nc * scb_n] = f2bf(v);
        }
    }
}

// Fused Linformer attention per (b, h, 16-row tile):
//   dots = q_tile(16x64) @ kv^T(64x256) * scale -> LDS; softmax; out = attn @ kv
//   written in-place over q.  block = (32,8).
#define SMP (KPROJ + 4)   // +4 floats row pad: 16 fragment lanes hit 16 distinct banks
__global__ void k_attn(__bf16* __restrict__ q, const __bf16* __restrict__ kv,
                       const __bf16* __restrict__ kvT, float scale) {
    __shared__ float sm[16][SMP];
    __shared__ float red[16][16];
    const int lane = threadIdx.x, wave = threadIdx.y;
    const int nt = blockIdx.x, hh = blockIdx.y, b = blockIdx.z;

    __bf16* qbase = q + ((long)(b * SEQ + nt * 16) * DIM + hh * DHD);  // row stride DIM
    const __bf16* kvb  = kv  + (long)b * (KPROJ * DHD);  // (K, DHD) row-major: d contiguous
    const __bf16* kvTb = kvT + (long)b * (KPROJ * DHD);  // (DHD, K) row-major: k contiguous

    // stage 1: scores; each wave -> 2 column tiles of 16 kv-rows
    v8f acc[2] = {};
#pragma unroll
    for (int k0 = 0; k0 < DHD; k0 += 32) {
        v16bf a = load_a_rm(qbase + k0, DIM, lane);
#pragma unroll
        for (int t = 0; t < 2; ++t) {
            int ncol = (wave * 2 + t) * 16;
            // B(k=d, n=kcol) = kv[kcol*DHD + d] -> col stride DHD, k contiguous
            v16bf bf = load_b_kc(kvb + (long)ncol * DHD + k0, DHD, lane);
            acc[t] = WMMA_BF16(a, bf, acc[t]);
        }
    }
    const int nlane = lane & 15, mh = (lane >> 4) * 8;
#pragma unroll
    for (int t = 0; t < 2; ++t)
#pragma unroll
        for (int r = 0; r < 8; ++r)
            sm[mh + r][(wave * 2 + t) * 16 + nlane] = acc[t][r] * scale;
    __syncthreads();

    // softmax over 256 cols: 16 threads per row, 16 cols each
    const int tid = wave * 32 + lane;
    const int row = tid >> 4, c16 = tid & 15;
    float mx = -3.0e38f;
#pragma unroll
    for (int j = 0; j < 16; ++j) mx = fmaxf(mx, sm[row][c16 + j * 16]);
    red[row][c16] = mx;
    __syncthreads();
    float m2 = -3.0e38f;
#pragma unroll
    for (int j = 0; j < 16; ++j) m2 = fmaxf(m2, red[row][j]);
    float ex[16], ssum = 0.0f;
#pragma unroll
    for (int j = 0; j < 16; ++j) { ex[j] = expf(sm[row][c16 + j * 16] - m2); ssum += ex[j]; }
    __syncthreads();
    red[row][c16] = ssum;
    __syncthreads();
    float tot = 0.0f;
#pragma unroll
    for (int j = 0; j < 16; ++j) tot += red[row][j];
    float inv = 1.0f / tot;
#pragma unroll
    for (int j = 0; j < 16; ++j) sm[row][c16 + j * 16] = ex[j] * inv;
    __syncthreads();

    // stage 2: out = attn(16x256) @ kv(256x64); waves 0..3 take one 16-col tile
    if (wave < 4) {
        v8f o = {};
        const int g = lane >> 4, mrow = lane & 15;
        for (int k0 = 0; k0 < KPROJ; k0 += 32) {
            // A-frag from LDS: two contiguous runs of 8 floats -> float4 pairs
            const float4* p0 = (const float4*)&sm[mrow][k0 + g * 8];
            const float4* p1 = (const float4*)&sm[mrow][k0 + 16 + g * 8];
            float4 f0 = p0[0], f1 = p0[1], f2 = p1[0], f3 = p1[1];
            v16bf a;
            a[0] = f2bf(f0.x);  a[1] = f2bf(f0.y);  a[2]  = f2bf(f0.z);  a[3]  = f2bf(f0.w);
            a[4] = f2bf(f1.x);  a[5] = f2bf(f1.y);  a[6]  = f2bf(f1.z);  a[7]  = f2bf(f1.w);
            a[8] = f2bf(f2.x);  a[9] = f2bf(f2.y);  a[10] = f2bf(f2.z);  a[11] = f2bf(f2.w);
            a[12] = f2bf(f3.x); a[13] = f2bf(f3.y); a[14] = f2bf(f3.z); a[15] = f2bf(f3.w);
            // B(k,n=d) = kvT[d*KPROJ + k] -> col stride KPROJ, k contiguous
            v16bf bf = load_b_kc(kvTb + (long)(wave * 16) * KPROJ + k0, KPROJ, lane);
            o = WMMA_BF16(a, bf, o);
        }
#pragma unroll
        for (int r = 0; r < 8; ++r)
            qbase[(long)(mh + r) * DIM + wave * 16 + nlane] = f2bf(o[r]);
    }
}

// Classifier: out(16,10) = h[:,0,:] @ Wcls + bcls, scalar fp32
__global__ void k_cls(const float* __restrict__ h, const float* __restrict__ Wcls,
                      const float* __restrict__ bcls, float* __restrict__ out) {
    int t = threadIdx.x;
    if (t >= BB * NCLS) return;
    int b = t / NCLS, c = t % NCLS;
    const float* row = h + (long)b * SEQ * DIM;  // token 0
    float s = bcls[c];
    for (int d = 0; d < DIM; ++d) s += row[d] * Wcls[d * NCLS + c];
    out[t] = s;
}

// ---------------- host ----------------

static inline void gemm(hipStream_t s, const __bf16* A, const __bf16* B,
                        const float* bias, const float* resid, float* Cf, __bf16* Cb,
                        int M, int Nn, int Kk,
                        long sa_row, long sa_b, long sb_col, long sb_b,
                        int ldc, long scb_m, long scb_n, long sc_b, int gelu, int batch) {
    dim3 g(Nn / 64, M / 64, batch), blk(32, 4, 1);
    k_gemm<<<g, blk, 0, s>>>(A, B, bias, resid, Cf, Cb, M, Kk,
                             sa_row, sa_b, sb_col, sb_b,
                             ldc, scb_m, scb_n, sc_b, gelu);
}

extern "C" void kernel_launch(void* const* d_in, const int* in_sizes, int n_in,
                              void* d_out, int out_size, void* d_ws, size_t ws_size,
                              hipStream_t stream) {
    (void)in_sizes; (void)n_in; (void)out_size; (void)ws_size;
    const int*   x     = (const int*)  d_in[0];
    const float* emb   = (const float*)d_in[1];
    const float* pos   = (const float*)d_in[2];
    const float* ln1_g = (const float*)d_in[3];
    const float* ln1_b = (const float*)d_in[4];
    const float* Wq    = (const float*)d_in[5];
    const float* Wk    = (const float*)d_in[6];
    const float* Ek    = (const float*)d_in[7];
    const float* Wo    = (const float*)d_in[8];
    const float* bo    = (const float*)d_in[9];
    const float* ln2_g = (const float*)d_in[10];
    const float* ln2_b = (const float*)d_in[11];
    const float* W1    = (const float*)d_in[12];
    const float* b1    = (const float*)d_in[13];
    const float* W2    = (const float*)d_in[14];
    const float* b2    = (const float*)d_in[15];
    const float* Wcls  = (const float*)d_in[16];
    const float* bcls  = (const float*)d_in[17];
    float* out = (float*)d_out;

    // workspace carve-up
    char* ws = (char*)d_ws;
    float*  h    = (float*)ws;  ws += (size_t)BB * SEQ * DIM * 4;
    __bf16* y    = (__bf16*)ws; ws += (size_t)BB * SEQ * DIM * 2;
    __bf16* q    = (__bf16*)ws; ws += (size_t)BB * SEQ * DIM * 2;     // q, then attn output
    __bf16* tmpT = (__bf16*)ws; ws += (size_t)DHD * BB * SEQ * 2;     // (y@Wk)^T : (DHD, B*SEQ)
    __bf16* kvb  = (__bf16*)ws; ws += (size_t)BB * KPROJ * DHD * 2;   // (B,K,DHD)
    __bf16* kvT  = (__bf16*)ws; ws += (size_t)BB * DHD * KPROJ * 2;   // (B,DHD,K)
    __bf16* hid  = (__bf16*)ws; ws += (size_t)CHROWS * FFDIM * 2;     // FFN chunk
    __bf16* wqT  = (__bf16*)ws; ws += (size_t)NLAYER * DIM * DIM * 2;
    __bf16* wkT  = (__bf16*)ws; ws += (size_t)NLAYER * DHD * DIM * 2;
    __bf16* ekT  = (__bf16*)ws; ws += (size_t)NLAYER * KPROJ * SEQ * 2;
    __bf16* woT  = (__bf16*)ws; ws += (size_t)NLAYER * DIM * DIM * 2;
    __bf16* w1T  = (__bf16*)ws; ws += (size_t)NLAYER * FFDIM * DIM * 2;
    __bf16* w2T  = (__bf16*)ws; ws += (size_t)NLAYER * DIM * FFDIM * 2;

    const long rows = (long)BB * SEQ;  // 65536

    // weights: fp32 -> bf16, transposed so B-fragments are k-contiguous
    auto cvtT = [&](const float* src, __bf16* dst, int R, int C) {
        long n = (long)NLAYER * R * C;
        k_cvt_t<<<(unsigned)((n + 255) / 256), 256, 0, stream>>>(src, dst, R, C, n);
    };
    cvtT(Wq, wqT, DIM, DIM);      // (DIM,DIM)   -> col stride DIM
    cvtT(Wk, wkT, DIM, DHD);      // (DHD,DIM)   -> col stride DIM
    cvtT(Ek, ekT, SEQ, KPROJ);    // (KPROJ,SEQ) -> row-major A, row stride SEQ
    cvtT(Wo, woT, DIM, DIM);
    cvtT(W1, w1T, DIM, FFDIM);    // (FFDIM,DIM) -> col stride DIM
    cvtT(W2, w2T, FFDIM, DIM);    // (DIM,FFDIM) -> col stride FFDIM

    // embedding + posenc
    {
        long total = rows * DIM;
        k_embed<<<(unsigned)((total + 255) / 256), 256, 0, stream>>>(x, emb, pos, h);
    }

    for (int l = 0; l < NLAYER; ++l) {
        // ---- attention ----
        k_ln<<<(unsigned)rows, 256, 0, stream>>>(h, ln1_g + l * DIM, ln1_b + l * DIM, y);

        // q = y @ Wq[l]  (65536x512 @ 512x512) -> bf16 row-major
        gemm(stream, y, wqT + (long)l * DIM * DIM, nullptr, nullptr, nullptr, q,
             (int)rows, DIM, DIM, DIM, 0, DIM, 0, DIM, DIM, 1, 0, 0, 1);

        // tmpT = (y @ Wk[l])^T  (65536x512 @ 512x64) stored (DHD, rows)
        gemm(stream, y, wkT + (long)l * DHD * DIM, nullptr, nullptr, nullptr, tmpT,
             (int)rows, DHD, DIM, DIM, 0, DIM, 0, DHD, 1, rows, 0, 0, 1);

        // kv[b] = Ek[l]^T @ tmp[b]  (256x4096 @ 4096x64): A=ekT row-major, B=tmpT k-contig
        gemm(stream, ekT + (long)l * KPROJ * SEQ, tmpT, nullptr, nullptr, nullptr, kvb,
             KPROJ, DHD, SEQ, SEQ, 0, rows, SEQ,
             DHD, DHD, 1, (long)KPROJ * DHD, 0, BB);

        // kvT for the attn@kv stage
        {
            long n = (long)BB * KPROJ * DHD;
            k_kvT<<<(unsigned)((n + 255) / 256), 256, 0, stream>>>(kvb, kvT);
        }

        // fused dots -> softmax -> attn@kv (in-place over q)
        {
            dim3 g(SEQ / 16, NH, BB), blk(32, 8, 1);
            k_attn<<<g, blk, 0, stream>>>(q, kvb, kvT, 0.125f /* DH^-0.5 */);
        }

        // h = h + out @ Wo[l] + bo[l]
        gemm(stream, q, woT + (long)l * DIM * DIM, bo + l * DIM, h, h, nullptr,
             (int)rows, DIM, DIM, DIM, 0, DIM, 0, DIM, 0, 0, 0, 0, 1);

        // ---- feed-forward ----
        k_ln<<<(unsigned)rows, 256, 0, stream>>>(h, ln2_g + l * DIM, ln2_b + l * DIM, y);

        for (long r0 = 0; r0 < rows; r0 += CHROWS) {
            // hid = gelu(y_chunk @ W1[l] + b1[l]) -> bf16 row-major
            gemm(stream, y + r0 * DIM, w1T + (long)l * FFDIM * DIM,
                 b1 + l * FFDIM, nullptr, nullptr, hid,
                 CHROWS, FFDIM, DIM, DIM, 0, DIM, 0, FFDIM, FFDIM, 1, 0, 1, 1);
            // h_chunk += hid @ W2[l] + b2[l]
            gemm(stream, hid, w2T + (long)l * DIM * FFDIM,
                 b2 + l * DIM, h + r0 * DIM, h + r0 * DIM, nullptr,
                 CHROWS, DIM, FFDIM, FFDIM, 0, FFDIM, 0, DIM, 0, 0, 0, 0, 1);
        }
    }

    // classifier on token 0
    k_cls<<<1, 256, 0, stream>>>(h, Wcls, bcls, out);
}
